// Sub_31318901522624
// MI455X (gfx1250) — matrix operand order repaired and checked by reading, not compiled, not run on hardware
//
#include <hip/hip_runtime.h>

// CDNA5 / gfx1250, wave32. WMMA shapes & VGPR layouts per CDNA5 ISA.
typedef __attribute__((ext_vector_type(16))) __bf16 v16bf;
typedef __attribute__((ext_vector_type(8)))  __bf16 v8bf;
typedef __attribute__((ext_vector_type(4)))  __bf16 v4bf;
typedef __attribute__((ext_vector_type(2)))  __bf16 bf16x2;
typedef __attribute__((ext_vector_type(8)))  float  v8f;

#define LBITS 32   // digit positions
#define KDIG  16   // radix

// One wave = one 16-batch tile, scanned serially over L.
// res[b,l,v] = c0*q[v] + c1*q[(v+1)&15],  q = circ-corr(p1,p2) via one-hot-table GEMM
// (bf16 in / f32 accumulate on v_wmma_f32_16x16x32_bf16). Borrow scan kept fp32-exact.
__global__ __launch_bounds__(256) void borrow_sub_scan_wmma(
    const float* __restrict__ op1,
    const float* __restrict__ op2,
    float* __restrict__ out,
    int nBatch)
{
  const int lane   = threadIdx.x & 31;
  const int waveId = blockIdx.x * (blockDim.x >> 5) + (threadIdx.x >> 5);
  const int b0     = waveId << 4;
  if (b0 + 16 > nBatch) return;           // uniform per wave: EXEC stays all-ones

  const int m     = lane & 15;            // batch row owned by this lane (A layout)
  const int khalf = lane >> 4;            // K-half of the wave
  const int col   = lane & 15;            // output column (B/C layout)

  // ---- constant one-hot table, 8 B-fragments (K=256 in chunks of 32), built once ----
  // B 32x16 layout matches A: slot e -> K=(e<8?e:e+8)+8*khalf,
  // row z=(x,y): x=2kc+(e>>3), y=(e&7)+8*khalf.  Entry = ((x-y)&15 == col).
  v16bf bfrag[8];
  #pragma unroll
  for (int kc = 0; kc < 8; ++kc) {
    #pragma unroll
    for (int e = 0; e < 16; ++e) {
      const int x = 2 * kc + (e >> 3);
      const int y = (e & 7) + 8 * khalf;
      bfrag[kc][e] = (((x - y) & 15) == col) ? (__bf16)1.0f : (__bf16)0.0f;
    }
  }

  const float* p1base = op1 + (size_t)(b0 + m) * (LBITS * KDIG);
  const float* p2base = op2 + (size_t)(b0 + m) * (LBITS * KDIG);

  float c0 = 1.0f, c1 = 0.0f;             // borrow-in distribution, fp32-exact

  for (int l = 0; l < LBITS; ++l) {
    // ---- stream this step's digit rows (float4 = GLOBAL_LOAD_B128) ----
    float p1r[16], p2r[16];
    {
      const float4* a4 = (const float4*)(p1base + l * KDIG);
      const float4* b4 = (const float4*)(p2base + l * KDIG);
      #pragma unroll
      for (int j = 0; j < 4; ++j) {
        float4 t = a4[j];
        p1r[4*j+0]=t.x; p1r[4*j+1]=t.y; p1r[4*j+2]=t.z; p1r[4*j+3]=t.w;
        float4 u = b4[j];
        p2r[4*j+0]=u.x; p2r[4*j+1]=u.y; p2r[4*j+2]=u.z; p2r[4*j+3]=u.w;
      }
    }
    // speculative prefetch of next step's rows (dropped harmlessly past the end)
    __builtin_prefetch(p1base + (l + 1) * KDIG, 0, 1);
    __builtin_prefetch(p2base + (l + 1) * KDIG, 0, 1);

    // ---- fp32 borrow statistics: s0 = sum_{x<y} p1[x]p2[y], s1 = s0 + <p1,p2> ----
    float cum = 0.f, s0 = 0.f, dotp = 0.f, S2 = 0.f;
    #pragma unroll
    for (int y = 0; y < 16; ++y) {
      s0   += p2r[y] * cum;
      cum  += p1r[y];
      dotp += p1r[y] * p2r[y];
      S2   += p2r[y];
    }
    const float T  = cum * S2;            // total joint mass factor (S1*S2)
    const float s1 = s0 + dotp;

    // ---- packed bf16 staging (v_cvt_pk_bf16_f32) ----
    // A 16x32 layout slot e -> K=(e<8?e:e+8)+8*khalf => x = 2kc+(e>>3), y=(e&7)+8*khalf.
    // Slot pair (2g,2g+1): same x, consecutive y -> natural bf16x2 packing.
    bf16x2 p1bb[16];                      // splat pairs of p1 digits
    #pragma unroll
    for (int x = 0; x < 16; ++x) {
      const __bf16 h = (__bf16)p1r[x];
      p1bb[x][0] = h; p1bb[x][1] = h;
    }
    bf16x2 p2s2[8];                       // p2 slot-pairs in A order
    #pragma unroll
    for (int g = 0; g < 8; ++g) {
      const int yb = ((g < 4) ? 2 * g : 2 * g - 8) + khalf * 8;
      p2s2[g][0] = (__bf16)p2r[yb];
      p2s2[g][1] = (__bf16)p2r[yb + 1];
    }

    // ---- q tile via 8 chained v_wmma_f32_16x16x32_bf16 (K=256 outer-product GEMM) ----
    v8f acc = {};
    #pragma unroll
    for (int kc = 0; kc < 8; ++kc) {
      // packed products: v_pk_mul_bf16 (2 A-elements per instruction)
      bf16x2 pr0 = p1bb[2*kc]     * p2s2[0];
      bf16x2 pr1 = p1bb[2*kc]     * p2s2[1];
      bf16x2 pr2 = p1bb[2*kc]     * p2s2[2];
      bf16x2 pr3 = p1bb[2*kc]     * p2s2[3];
      bf16x2 pr4 = p1bb[2*kc + 1] * p2s2[4];
      bf16x2 pr5 = p1bb[2*kc + 1] * p2s2[5];
      bf16x2 pr6 = p1bb[2*kc + 1] * p2s2[6];
      bf16x2 pr7 = p1bb[2*kc + 1] * p2s2[7];
      // assemble the 16-element A fragment (register coalescing only)
      v4bf  q01 = __builtin_shufflevector(pr0, pr1, 0, 1, 2, 3);
      v4bf  q23 = __builtin_shufflevector(pr2, pr3, 0, 1, 2, 3);
      v4bf  q45 = __builtin_shufflevector(pr4, pr5, 0, 1, 2, 3);
      v4bf  q67 = __builtin_shufflevector(pr6, pr7, 0, 1, 2, 3);
      v8bf  h0  = __builtin_shufflevector(q01, q23, 0, 1, 2, 3, 4, 5, 6, 7);
      v8bf  h1  = __builtin_shufflevector(q45, q67, 0, 1, 2, 3, 4, 5, 6, 7);
      v16bf af  = __builtin_shufflevector(h0, h1, 0, 1, 2, 3, 4, 5, 6, 7,
                                          8, 9, 10, 11, 12, 13, 14, 15);
      acc = __builtin_amdgcn_wmma_f32_16x16x32_bf16(
          false, af, false, bfrag[kc], (short)0, acc, false, false);
    }

    // ---- combine with borrow-in (uses PRE-update c) and store ----
    // C layout: lanes 0-15 hold rows 0..7, lanes 16-31 rows 8..15, col = lane&15.
    const int srcRot = (lane & 16) | ((lane + 1) & 15);   // q[., (col+1)&15]
    #pragma unroll
    for (int r = 0; r < 8; ++r) {
      const float q   = acc[r];
      const float qp1 = __shfl(q, srcRot, 32);
      const int   mq  = khalf * 8 + r;                    // batch row of acc[r]
      const float c0b = __shfl(c0, mq, 32);               // c lives on lane mq (<16)
      const float c1b = __shfl(c1, mq, 32);
      out[((size_t)(b0 + mq) * LBITS + l) * KDIG + col] = c0b * q + c1b * qp1;
    }

    // ---- exact fp32 borrow scan update ----
    const float c1n = c0 * s0 + c1 * s1;
    c0 = T * (c0 + c1) - c1n;
    c1 = c1n;
  }
}

extern "C" void kernel_launch(void* const* d_in, const int* in_sizes, int n_in,
                              void* d_out, int out_size, void* d_ws, size_t ws_size,
                              hipStream_t stream) {
  const float* op1 = (const float*)d_in[0];
  const float* op2 = (const float*)d_in[1];
  float* out = (float*)d_out;

  const int nBatch = in_sizes[0] / (LBITS * KDIG);   // 65536
  const int tiles  = nBatch >> 4;                    // one wave per 16 batches
  const int wavesPerBlock = 8;                       // 256 threads, wave32
  const int blocks = (tiles + wavesPerBlock - 1) / wavesPerBlock;

  borrow_sub_scan_wmma<<<blocks, wavesPerBlock * 32, 0, stream>>>(op1, op2, out, nBatch);
}
